// MultiHeadAttention_42898133353006
// MI455X (gfx1250) — compile-verified, matrix operand order checked by reading, not run on hardware
//
#include <hip/hip_runtime.h>
#include <hip/hip_bf16.h>

// ---------------------------------------------------------------------------
// MI455X (gfx1250) fused MHA forward:
//   fp32 inputs -> bf16 staging -> v_wmma_f32_16x16x32_bf16 everywhere,
//   fp32 softmax + fp32 accumulation, fp32 output.
// ---------------------------------------------------------------------------

typedef __bf16 bf16;
typedef __attribute__((ext_vector_type(8)))  __bf16 bf16x8;
typedef __attribute__((ext_vector_type(16))) __bf16 bf16x16;
typedef __attribute__((ext_vector_type(8)))  float  floatx8;

static constexpr int kN  = 2;
static constexpr int kS  = 2048;
static constexpr int kD  = 1024;
static constexpr int kH  = 16;
static constexpr int kHd = 64;

__device__ __forceinline__ floatx8 zero8() {
  floatx8 z;
#pragma unroll
  for (int i = 0; i < 8; ++i) z[i] = 0.0f;
  return z;
}

// Load one 16-element bf16 WMMA A/B fragment for this lane.
// p points at this lane's row start (already offset by the K-base).
// Packing per ISA 7.12.2 (16-bit A, 16x32):
//   lanes 0-15 : K = {0..7, 16..23}   -> bytes {0..15, 32..47}
//   lanes 16-31: K = {8..15, 24..31}  -> bytes {16..31, 48..63}
__device__ __forceinline__ bf16x16 load_frag(const bf16* p, int half) {
  const bf16* q = p + half * 8;
  bf16x8 lo = *(const bf16x8*)(q);
  bf16x8 hi = *(const bf16x8*)(q + 16);
  return __builtin_shufflevector(lo, hi, 0, 1, 2, 3, 4, 5, 6, 7,
                                 8, 9, 10, 11, 12, 13, 14, 15);
}

__device__ __forceinline__ floatx8 wmma_bf16(bf16x16 a, bf16x16 b, floatx8 c) {
  return __builtin_amdgcn_wmma_f32_16x16x32_bf16(
      /*neg_a=*/false, a, /*neg_b=*/false, b,
      /*c_mod=*/(short)0, c, /*reuse_a=*/false, /*reuse_b=*/false);
}

// ---------------------------------------------------------------------------
// fp32 -> bf16 elementwise convert
// ---------------------------------------------------------------------------
__global__ __launch_bounds__(256) void cvt_f32_bf16_kernel(
    const float* __restrict__ in, bf16* __restrict__ out, int n) {
  int idx = blockIdx.x * 256 + threadIdx.x;
  if (idx < n) out[idx] = (bf16)in[idx];
}

// ---------------------------------------------------------------------------
// Transpose + convert: Wt[n][k] = (bf16) W[k][n], D x D, LDS tiled (coalesced
// on both sides).
// ---------------------------------------------------------------------------
__global__ __launch_bounds__(256) void transpose_cvt_kernel(
    const float* __restrict__ W, bf16* __restrict__ Wt) {
  __shared__ float tile[32][33];
  int tx = threadIdx.x, ty = threadIdx.y;
  int x = blockIdx.x * 32 + tx;  // input column
  int y0 = blockIdx.y * 32;      // input row base
#pragma unroll
  for (int i = 0; i < 4; ++i) {
    int k = y0 + ty + i * 8;
    tile[ty + i * 8][tx] = W[(size_t)k * kD + x];
  }
  __syncthreads();
#pragma unroll
  for (int i = 0; i < 4; ++i) {
    int n = blockIdx.x * 32 + ty + i * 8;  // output row (= input column)
    int k = y0 + tx;                       // output col (= input row)
    Wt[(size_t)n * kD + k] = (bf16)tile[tx][ty + i * 8];
  }
}

// ---------------------------------------------------------------------------
// Wave-level WMMA GEMM: Y = A[M x K] * Bt^T + bias, bf16 in / fp32 acc.
// Each wave owns a 16(M) x 64(N) tile; A fragment reused across 4 N-tiles.
//  mode 0: bf16 head-split store  dst[n][h][s][hd], value = (acc+bias)*scale
//  mode 1: bf16 transposed store  dst[n][h][hd][s]  (for V)
//  mode 2: fp32 row-major store   dst[row][col]     (final projection)
// ---------------------------------------------------------------------------
__global__ __launch_bounds__(256) void gemm_wmma_kernel(
    const bf16* __restrict__ A, const bf16* __restrict__ Bt,
    const float* __restrict__ bias, void* __restrict__ out,
    int M, int K, int Nout, int mode, float scale) {
  int wave = blockIdx.x * (blockDim.x >> 5) + (threadIdx.x >> 5);
  int lane = threadIdx.x & 31;
  int half = lane >> 4;
  int lr = lane & 15;
  int tilesN = Nout >> 6;
  int m0 = (wave / tilesN) << 4;
  int n0 = (wave % tilesN) << 6;
  if (m0 >= M) return;

  const bf16* arow = A + (size_t)(m0 + lr) * K;
  const bf16* brow0 = Bt + (size_t)(n0 + 0 * 16 + lr) * K;
  const bf16* brow1 = Bt + (size_t)(n0 + 1 * 16 + lr) * K;
  const bf16* brow2 = Bt + (size_t)(n0 + 2 * 16 + lr) * K;
  const bf16* brow3 = Bt + (size_t)(n0 + 3 * 16 + lr) * K;

  floatx8 acc[4];
#pragma unroll
  for (int t = 0; t < 4; ++t) acc[t] = zero8();

  for (int k = 0; k < K; k += 32) {
    __builtin_prefetch(arow + k + 128, 0, 3);
    bf16x16 av = load_frag(arow + k, half);
    acc[0] = wmma_bf16(av, load_frag(brow0 + k, half), acc[0]);
    acc[1] = wmma_bf16(av, load_frag(brow1 + k, half), acc[1]);
    acc[2] = wmma_bf16(av, load_frag(brow2 + k, half), acc[2]);
    acc[3] = wmma_bf16(av, load_frag(brow3 + k, half), acc[3]);
  }

#pragma unroll
  for (int t = 0; t < 4; ++t) {
    int col = n0 + t * 16 + lr;
    float bc = bias ? bias[col] : 0.0f;
#pragma unroll
    for (int r = 0; r < 8; ++r) {
      int row = m0 + r + half * 8;
      float v = (acc[t][r] + bc) * scale;
      if (mode == 2) {
        ((float*)out)[(size_t)row * Nout + col] = v;
      } else {
        int n = row >> 11;        // row = n*S + s, S = 2048
        int s = row & (kS - 1);
        int h = col >> 6;         // col = h*Hd + hd, Hd = 64
        int hd = col & (kHd - 1);
        size_t idx;
        if (mode == 0)
          idx = ((((size_t)n * kH + h) * kS + s) * kHd + hd);
        else
          idx = ((((size_t)n * kH + h) * kHd + hd) * kS + s);
        ((bf16*)out)[idx] = (bf16)v;
      }
    }
  }
}

// ---------------------------------------------------------------------------
// Flash-style attention: one wave per (n, h, 16-query block).
//  Qb/Kb: [N][H][S][Hd] bf16 (Q pre-scaled by 1/sqrt(Hd))
//  Vt:    [N][H][Hd][S] bf16
//  Ob:    [N][S][D]     bf16 (concatenated heads, input to final projection)
// Streams 32 keys per step: 4 WMMAs for scores, online softmax in fp32,
// P transposed to A-layout through LDS, 4 WMMAs for P*V.
// ---------------------------------------------------------------------------
__global__ __launch_bounds__(32) void attn_kernel(
    const bf16* __restrict__ Qb, const bf16* __restrict__ Kb,
    const bf16* __restrict__ Vt, bf16* __restrict__ Ob) {
  __shared__ __align__(16) bf16 Pl[16][32];

  int bid = blockIdx.x;
  int qblk = bid & 127;            // S/16 = 128 query blocks
  int h = (bid >> 7) & (kH - 1);
  int n = bid >> 11;
  int lane = threadIdx.x & 31;
  int half = lane >> 4;
  int lr = lane & 15;

  size_t headOff = (size_t)(n * kH + h) * kS * kHd;
  const bf16* qrow = Qb + headOff + (size_t)(qblk * 16 + lr) * kHd;
  bf16x16 qa0 = load_frag(qrow, half);        // K-dim 0..31 of Hd
  bf16x16 qa1 = load_frag(qrow + 32, half);   // K-dim 32..63 of Hd
  const bf16* kbase = Kb + headOff;
  const bf16* vbase = Vt + headOff;           // [Hd][S]

  float mrow[8], lsum[8];
  floatx8 Oacc[4];
#pragma unroll
  for (int t = 0; t < 4; ++t) Oacc[t] = zero8();
#pragma unroll
  for (int r = 0; r < 8; ++r) { mrow[r] = -3.0e38f; lsum[r] = 0.0f; }

  for (int j = 0; j < kS; j += 32) {
    const bf16* k0row = kbase + (size_t)(j + lr) * kHd;
    const bf16* k1row = kbase + (size_t)(j + 16 + lr) * kHd;
    floatx8 s0 = zero8(), s1 = zero8();
    s0 = wmma_bf16(qa0, load_frag(k0row, half), s0);
    s0 = wmma_bf16(qa1, load_frag(k0row + 32, half), s0);
    s1 = wmma_bf16(qa0, load_frag(k1row, half), s1);
    s1 = wmma_bf16(qa1, load_frag(k1row + 32, half), s1);

    float p0[8], p1[8], sc[8];
#pragma unroll
    for (int r = 0; r < 8; ++r) {
      // Row-wise max: row lives across 16 lanes of one half; xor masks 1/2/4/8
      // reduce within each half independently.
      float mx = fmaxf(s0[r], s1[r]);
      mx = fmaxf(mx, __shfl_xor(mx, 1, 32));
      mx = fmaxf(mx, __shfl_xor(mx, 2, 32));
      mx = fmaxf(mx, __shfl_xor(mx, 4, 32));
      mx = fmaxf(mx, __shfl_xor(mx, 8, 32));
      float mnew = fmaxf(mrow[r], mx);
      sc[r] = __expf(mrow[r] - mnew);
      mrow[r] = mnew;
      p0[r] = __expf(s0[r] - mnew);
      p1[r] = __expf(s1[r] - mnew);
      lsum[r] = lsum[r] * sc[r] + p0[r] + p1[r];
    }
#pragma unroll
    for (int t = 0; t < 4; ++t)
#pragma unroll
      for (int r = 0; r < 8; ++r) Oacc[t][r] *= sc[r];

    // Transpose P (C-layout -> row-major) through LDS, then reload in A-layout.
#pragma unroll
    for (int r = 0; r < 8; ++r) {
      Pl[r + half * 8][lr] = (bf16)p0[r];
      Pl[r + half * 8][lr + 16] = (bf16)p1[r];
    }
    __syncthreads();
    bf16x16 pa = load_frag(&Pl[lr][0], half);
    __syncthreads();

#pragma unroll
    for (int t = 0; t < 4; ++t) {
      const bf16* vrow = vbase + (size_t)(t * 16 + lr) * kS + j;
      Oacc[t] = wmma_bf16(pa, load_frag(vrow, half), Oacc[t]);
    }
  }

  float inv[8];
#pragma unroll
  for (int r = 0; r < 8; ++r) {
    float l = lsum[r];
    l += __shfl_xor(l, 1, 32);
    l += __shfl_xor(l, 2, 32);
    l += __shfl_xor(l, 4, 32);
    l += __shfl_xor(l, 8, 32);
    inv[r] = 1.0f / l;
  }
#pragma unroll
  for (int t = 0; t < 4; ++t)
#pragma unroll
    for (int r = 0; r < 8; ++r) {
      int s = qblk * 16 + r + half * 8;
      int d = h * kHd + t * 16 + lr;
      Ob[((size_t)n * kS + s) * kD + d] = (bf16)(Oacc[t][r] * inv[r]);
    }
}

// ---------------------------------------------------------------------------
extern "C" void kernel_launch(void* const* d_in, const int* in_sizes, int n_in,
                              void* d_out, int out_size, void* d_ws, size_t ws_size,
                              hipStream_t stream) {
  const float* query = (const float*)d_in[0];
  const float* key   = (const float*)d_in[1];
  const float* value = (const float*)d_in[2];
  const float* Wq = (const float*)d_in[3];
  const float* bq = (const float*)d_in[4];
  const float* Wk = (const float*)d_in[5];
  const float* bk = (const float*)d_in[6];
  const float* Wv = (const float*)d_in[7];
  const float* bv = (const float*)d_in[8];
  const float* Wp = (const float*)d_in[9];
  const float* bp = (const float*)d_in[10];

  const int MS = kN * kS;  // 4096 rows
  char* ws = (char*)d_ws;
  size_t off = 0;
  auto take = [&](size_t bytes) -> char* {
    char* p = ws + off;
    off += (bytes + 255) & ~(size_t)255;
    return p;
  };

  bf16* Xq = (bf16*)take((size_t)MS * kD * 2);
  bf16* Xk = (bf16*)take((size_t)MS * kD * 2);
  bf16* Xv = (bf16*)take((size_t)MS * kD * 2);
  bf16* Wqt = (bf16*)take((size_t)kD * kD * 2);
  bf16* Wkt = (bf16*)take((size_t)kD * kD * 2);
  bf16* Wvt = (bf16*)take((size_t)kD * kD * 2);
  bf16* Wpt = (bf16*)take((size_t)kD * kD * 2);
  bf16* Qb  = (bf16*)take((size_t)MS * kD * 2);  // [N][H][S][Hd]
  bf16* Kb  = (bf16*)take((size_t)MS * kD * 2);  // [N][H][S][Hd]
  bf16* Vt  = (bf16*)take((size_t)MS * kD * 2);  // [N][H][Hd][S]
  bf16* Ob  = (bf16*)take((size_t)MS * kD * 2);  // [N][S][D]

  const int nElem = MS * kD;
  const int cvtBlocks = (nElem + 255) / 256;
  cvt_f32_bf16_kernel<<<cvtBlocks, 256, 0, stream>>>(query, Xq, nElem);
  cvt_f32_bf16_kernel<<<cvtBlocks, 256, 0, stream>>>(key, Xk, nElem);
  cvt_f32_bf16_kernel<<<cvtBlocks, 256, 0, stream>>>(value, Xv, nElem);

  dim3 tgrid(kD / 32, kD / 32), tblk(32, 8);
  transpose_cvt_kernel<<<tgrid, tblk, 0, stream>>>(Wq, Wqt);
  transpose_cvt_kernel<<<tgrid, tblk, 0, stream>>>(Wk, Wkt);
  transpose_cvt_kernel<<<tgrid, tblk, 0, stream>>>(Wv, Wvt);
  transpose_cvt_kernel<<<tgrid, tblk, 0, stream>>>(Wp, Wpt);

  const int waves = (MS / 16) * (kD / 64);
  const int gemmBlocks = (waves + 7) / 8;
  // Q: fold 1/sqrt(Hd)=0.125 into the stored Q
  gemm_wmma_kernel<<<gemmBlocks, 256, 0, stream>>>(Xq, Wqt, bq, Qb, MS, kD, kD, 0, 0.125f);
  gemm_wmma_kernel<<<gemmBlocks, 256, 0, stream>>>(Xk, Wkt, bk, Kb, MS, kD, kD, 0, 1.0f);
  gemm_wmma_kernel<<<gemmBlocks, 256, 0, stream>>>(Xv, Wvt, bv, Vt, MS, kD, kD, 1, 1.0f);

  attn_kernel<<<kN * kH * (kS / 16), 32, 0, stream>>>(Qb, Kb, Vt, Ob);

  gemm_wmma_kernel<<<gemmBlocks, 256, 0, stream>>>(Ob, Wpt, bp, (float*)d_out,
                                                   MS, kD, kD, 2, 1.0f);
}
